// RNNLayer_19628000543503
// MI455X (gfx1250) — matrix operand order repaired and checked by reading, not compiled, not run on hardware
//
#include <hip/hip_runtime.h>
#include <math.h>

// Problem constants (reference: B=64, S=512, D=1024, H=1024, L=2)
constexpr int kB   = 64;
constexpr int kS   = 512;
constexpr int kD   = 1024;
constexpr int kH   = 1024;
constexpr long kKtot = 2048;           // fused K = D + H (layer0) / H + H (layer1)
constexpr int kNT  = 64;               // column tiles of 16 (H/16)

typedef __attribute__((ext_vector_type(16))) __bf16 v16bf;
typedef __attribute__((ext_vector_type(8)))  __bf16 v8bf;
typedef __attribute__((ext_vector_type(8)))  float  v8f;
typedef __attribute__((ext_vector_type(4)))  float  f4;

// ---- fp32 <-> bf16 bits (round-to-nearest-even) ----
__device__ inline unsigned short f2bf(float f) {
    unsigned int u = __float_as_uint(f);
    unsigned int r = u + 0x7FFFu + ((u >> 16) & 1u);
    return (unsigned short)(r >> 16);
}
__device__ inline float bf2f(unsigned short u) {
    return __uint_as_float(((unsigned int)u) << 16);
}

// Pack [Wih ; Whh] (each [1024,1024] f32 row-major) into bf16 WMMA-B fragments.
// Block (kb, nb) = 32 lanes x 16 bf16; lane l holds W[kb*32 + (j<8 ? (l/16)*8 + j : 16 + (l/16)*8 + j-8)][nb*16 + l%16],
// j = 0..15 in the lane's 32-byte chunk (two B128 loads at consume time).
__global__ void pack_weights(const float* __restrict__ Wih,
                             const float* __restrict__ Whh,
                             unsigned short* __restrict__ Wp) {
    long e = (long)blockIdx.x * blockDim.x + threadIdx.x;      // 0 .. 2M-1
    if (e >= kKtot * kH) return;
    int blk    = (int)(e >> 9);      // /512 elems per block
    int within = (int)(e & 511);
    int lane   = within >> 4;
    int j      = within & 15;
    int kb     = blk >> 6;           // / kNT
    int nb     = blk & 63;
    int k      = kb * 32 + ((j < 8) ? ((lane >> 4) * 8 + j) : (16 + (lane >> 4) * 8 + (j - 8)));
    int n      = nb * 16 + (lane & 15);
    float f = (k < kD) ? Wih[(long)k * kH + n] : Whh[(long)(k - kD) * kH + n];
    Wp[e] = f2bf(f);
}

// One-time fp32 -> bf16 conversion of the whole input tensor (row-major preserved).
__global__ void cvt_x(const float* __restrict__ x, unsigned short* __restrict__ xb, long n) {
    long i = ((long)blockIdx.x * blockDim.x + threadIdx.x) * 8;
    if (i >= n) return;
    f4 a = *(const f4*)(x + i);
    f4 b = *(const f4*)(x + i + 4);
    v8bf r;
#pragma unroll
    for (int j = 0; j < 4; ++j) {
        r[j]     = (__bf16)a[j];
        r[4 + j] = (__bf16)b[j];
    }
    *(v8bf*)(xb + i) = r;
}

__global__ void zero_bf(unsigned short* __restrict__ p, int n) {
    int i = blockIdx.x * blockDim.x + threadIdx.x;
    if (i < n) p[i] = 0;
}

// A-fragment (16-bit A 16x32 layout): lane holds K = half*8 + 0..7 and 16+half*8 + 0..7.
__device__ inline v16bf load_a_frag(const unsigned short* __restrict__ p) {
    v8bf lo = *(const v8bf*)(p);        // 16 bytes
    v8bf hi = *(const v8bf*)(p + 16);   // 16 bytes
    return __builtin_shufflevector(lo, hi, 0,1,2,3,4,5,6,7,8,9,10,11,12,13,14,15);
}

// One layer-step: H_out[64,1024] = tanh( [A ; hPrev] @ Wp + bias )
// grid 64 blocks x 128 thr (4 waves). Block = one 16-wide column tile (waves share B),
// wave = one 16-row tile. K = 2048 in 64 blocks of 32 (first 32 from A, last 32 from hPrev).
// A and hPrev are bf16 row-major; accumulation + epilogue in fp32.
__global__ __launch_bounds__(128)
void rnn_step(const unsigned short* __restrict__ aBase, long aStride,
              const unsigned short* __restrict__ hPrev,
              const unsigned short* __restrict__ Wp,
              const float* __restrict__ bias,
              unsigned short* __restrict__ hOut,
              float* __restrict__ outBase, long outStride) {
    const int wave = threadIdx.x >> 5;
    const int lane = threadIdx.x & 31;
    const int mt   = wave;            // row tile 0..3
    const int nt   = blockIdx.x;      // col tile 0..63
    const int half = lane >> 4;
    const int m16  = lane & 15;
    const int row  = mt * 16 + m16;   // A-matrix source row (batch index)
    const int n    = nt * 16 + m16;   // output column for this lane

    const unsigned short* aRow  = aBase + (long)row * aStride + half * 8;
    const unsigned short* hRow  = hPrev + (long)row * kH      + half * 8;
    const unsigned short* wBase = Wp + ((long)nt << 9) + ((long)lane << 4);

    v8f acc = {};
#pragma unroll 4
    for (int kb = 0; kb < 32; ++kb) {            // K = 0..1023 : input activations
        v16bf af = load_a_frag(aRow + kb * 32);
        v16bf bf = *(const v16bf*)(wBase + ((long)kb * kNT << 9));
        acc = __builtin_amdgcn_wmma_f32_16x16x32_bf16(
                  false, af, false, bf, (short)0, acc, false, false);
    }
#pragma unroll 4
    for (int kb = 32; kb < 64; ++kb) {           // K = 1024..2047 : recurrent state
        v16bf af = load_a_frag(hRow + (kb - 32) * 32);
        v16bf bf = *(const v16bf*)(wBase + ((long)kb * kNT << 9));
        acc = __builtin_amdgcn_wmma_f32_16x16x32_bf16(
                  false, af, false, bf, (short)0, acc, false, false);
    }

    const float bn = bias[n];
#pragma unroll
    for (int i = 0; i < 8; ++i) {                // C/D layout: M = half*8 + i, N = lane%16
        int r = mt * 16 + half * 8 + i;
        float v = tanhf(acc[i] + bn);
        hOut[(long)r * kH + n] = f2bf(v);        // bf16 state for next step's A/hPrev
        if (outBase) outBase[(long)r * outStride + n] = v;  // fp32 output slab
    }
}

__global__ void copy_final(const unsigned short* __restrict__ h0,
                           const unsigned short* __restrict__ h1,
                           float* __restrict__ dst) {
    int i = blockIdx.x * blockDim.x + threadIdx.x;
    if (i < kB * kH) {
        dst[i]           = bf2f(h0[i]);
        dst[kB * kH + i] = bf2f(h1[i]);
    }
}

extern "C" void kernel_launch(void* const* d_in, const int* in_sizes, int n_in,
                              void* d_out, int out_size, void* d_ws, size_t ws_size,
                              hipStream_t stream) {
    (void)in_sizes; (void)n_in; (void)out_size; (void)ws_size;
    const float* x    = (const float*)d_in[0];
    const float* Wih0 = (const float*)d_in[1];
    const float* b0   = (const float*)d_in[2];
    const float* Whh0 = (const float*)d_in[3];
    const float* Wih1 = (const float*)d_in[4];
    const float* b1   = (const float*)d_in[5];
    const float* Whh1 = (const float*)d_in[6];
    float* out = (float*)d_out;

    // Workspace: W0p(4MB) | W1p(4MB) | x bf16 (64MB) | h0 ping-pong | h1 ping-pong
    char* ws = (char*)d_ws;
    unsigned short* W0p = (unsigned short*)ws;
    unsigned short* W1p = W0p + kKtot * kH;
    unsigned short* xbf = W1p + kKtot * kH;
    unsigned short* h0bf = xbf + (long)kB * kS * kD;
    unsigned short* h1bf = h0bf + 2 * kB * kH;

    // 1) Pack weights into bf16 WMMA-B fragment layout (L2-resident for all 512 steps).
    {
        long total = kKtot * kH;                          // 2M elements each
        int blocks = (int)((total + 255) / 256);
        pack_weights<<<blocks, 256, 0, stream>>>(Wih0, Whh0, W0p);
        pack_weights<<<blocks, 256, 0, stream>>>(Wih1, Whh1, W1p);
    }
    // 2) One-time x -> bf16 conversion (removes all cvt work from the recurrence).
    {
        long n = (long)kB * kS * kD;                      // 32M elements
        long threads = n / 8;
        cvt_x<<<(int)(threads / 256), 256, 0, stream>>>(x, xbf, n);
    }
    // 3) Zero hidden-state ping-pong buffers.
    {
        int total = 4 * kB * kH;
        zero_bf<<<(total + 255) / 256, 256, 0, stream>>>(h0bf, total);
    }
    // 4) Sequential scan: two layer-step GEMM kernels per timestep.
    for (int t = 0; t < kS; ++t) {
        int prev = t & 1;
        int cur  = (t + 1) & 1;
        // layer 0: A = x_t (bf16, base xbf + t*D, row stride S*D)
        rnn_step<<<kNT, 128, 0, stream>>>(
            xbf + (long)t * kD, (long)kS * kD,
            h0bf + (long)prev * kB * kH, W0p, b0,
            h0bf + (long)cur  * kB * kH,
            nullptr, 0);
        // layer 1: A = h0_t ; also writes all_outputs[b][t][:] in fp32
        rnn_step<<<kNT, 128, 0, stream>>>(
            h0bf + (long)cur * kB * kH, (long)kH,
            h1bf + (long)prev * kB * kH, W1p, b1,
            h1bf + (long)cur  * kB * kH,
            out + (long)t * kH, (long)kS * kH);
    }
    // 5) final_h [2, B, H] appended after all_outputs (t=511 wrote buffer index 0).
    copy_final<<<(kB * kH + 255) / 256, 256, 0, stream>>>(
        h0bf, h1bf, out + (long)kB * kS * kH);
}
